// DictNet_21912923144681
// MI455X (gfx1250) — matrix operand order repaired and checked by reading, not compiled, not run on hardware
//
#include <hip/hip_runtime.h>
#include <hip/hip_bf16.h>

#define NN 4096
#define FF 512
#define KK 11
#define NCLS 7
#define NG 128

typedef __attribute__((ext_vector_type(4)))  float  v4f;
typedef __attribute__((ext_vector_type(8)))  float  v8f;
typedef __attribute__((ext_vector_type(4)))  __bf16 v4bf;
typedef __attribute__((ext_vector_type(8)))  __bf16 v8bf;
typedef __attribute__((ext_vector_type(16))) __bf16 v16bf;

static __device__ __forceinline__ v16bf cat16(v8bf lo, v8bf hi) {
    return __builtin_shufflevector(lo, hi, 0,1,2,3,4,5,6,7,8,9,10,11,12,13,14,15);
}

// ---------------------------------------------------------------------------
// Kernel 1: normalize C, sparsity term, per-class masked counts, zero accums.
// scal layout: [0]=sparsity  [1]=hl1_acc  [2..8]=num[7]  [9..15]=cnt[7]
// ---------------------------------------------------------------------------
__global__ void prep_kernel(const float* __restrict__ C,
                            const unsigned char* __restrict__ mask,
                            const int* __restrict__ y,
                            float* __restrict__ scal, float* __restrict__ cn) {
    __shared__ float scnt[NCLS];
    int t = threadIdx.x;
    if (t < NCLS) scnt[t] = 0.0f;
    __syncthreads();
    for (int n = t; n < NN; n += blockDim.x)
        if (mask[n]) atomicAdd(&scnt[y[n]], 1.0f);
    __syncthreads();
    if (t == 0) {
        float c[KK], ss = 0.0f, as = 0.0f;
        for (int k = 0; k < KK; ++k) { c[k] = C[k]; ss += c[k] * c[k]; }
        float inv = 1.0f / sqrtf(ss);
        for (int k = 0; k < KK; ++k) { float v = c[k] * inv; cn[k] = v; as += fabsf(v); }
        float dim = sqrtf((float)KK);
        scal[0] = (dim - as) / (dim - 1.0f);   // ||Cn|| == 1
        scal[1] = 0.0f;
        for (int cc = 0; cc < NCLS; ++cc) { scal[2 + cc] = 0.0f; scal[9 + cc] = scnt[cc]; }
    }
}

// ---------------------------------------------------------------------------
// Kernel 2: L_hat = D @ Cn  -> bf16.  Dominant HBM stream (738 MB).
// 4 output elements per thread = 176 contiguous bytes = 11 aligned b128
// non-temporal loads (don't evict L2-resident WMMA operands).
// ---------------------------------------------------------------------------
__global__ __launch_bounds__(256) void lhat_kernel(const float* __restrict__ D,
                                                   const float* __restrict__ cn,
                                                   __bf16* __restrict__ lhat) {
    long long idx = (long long)blockIdx.x * blockDim.x + threadIdx.x;  // 0 .. NN*NN/4-1
    const v4f* p = (const v4f*)(D + idx * (4 * KK));

    float arr[4 * KK];
#pragma unroll
    for (int i = 0; i < KK; ++i) {
        v4f q = __builtin_nontemporal_load(p + i);
        arr[4 * i + 0] = q.x; arr[4 * i + 1] = q.y;
        arr[4 * i + 2] = q.z; arr[4 * i + 3] = q.w;
    }
    float c[KK];
#pragma unroll
    for (int k = 0; k < KK; ++k) c[k] = cn[k];

    v4bf out;
#pragma unroll
    for (int e = 0; e < 4; ++e) {
        float s = 0.0f;
#pragma unroll
        for (int k = 0; k < KK; ++k) s += arr[KK * e + k] * c[k];
        out[e] = (__bf16)s;
    }
    *(v4bf*)(lhat + idx * 4) = out;
}

// ---------------------------------------------------------------------------
// Kernel 2b: x [4096,512] f32 -> xT [512,4096] bf16 (tiled LDS transpose).
// Gives the GEMM's B-fragments K-contiguous rows => pure b128 loads, no LDS
// staging or barriers in the GEMM inner loop.
// ---------------------------------------------------------------------------
__global__ __launch_bounds__(256) void xpose_kernel(const float* __restrict__ x,
                                                    __bf16* __restrict__ xT) {
    __shared__ float tile[32][33];
    int tx = threadIdx.x & 31, ty = threadIdx.x >> 5;      // 32 x 8
    int n0 = blockIdx.x * 32, f0 = blockIdx.y * 32;
#pragma unroll
    for (int r = 0; r < 4; ++r) {
        int a = ty + r * 8;                                 // local n
        tile[a][tx] = x[(long long)(n0 + a) * FF + f0 + tx];
    }
    __syncthreads();
#pragma unroll
    for (int r = 0; r < 4; ++r) {
        int c = ty + r * 8;                                 // local f
        xT[(long long)(f0 + c) * NN + n0 + tx] = (__bf16)tile[tx][c];
    }
}

// ---------------------------------------------------------------------------
// Kernel 3: y_hat = x - L_hat @ x   (bf16 WMMA, f32 accumulate)
// block = 8 waves; block tile 128(M) x 64(N); wave = 16M x 64N, A reused 4x.
// A from lhat rows, B from xT rows: all contiguous b128, zero LDS/barriers.
// ---------------------------------------------------------------------------
__global__ __launch_bounds__(256) void yhat_gemm_kernel(const __bf16* __restrict__ lhat,
                                                        const __bf16* __restrict__ xT,
                                                        const float*  __restrict__ x,
                                                        float* __restrict__ yhat,
                                                        __bf16* __restrict__ yhb) {
    const int tid  = threadIdx.x;
    const int wave = tid >> 5;
    const int lane = tid & 31;
    const int m0   = blockIdx.x * 128 + wave * 16;
    const int n0   = blockIdx.y * 64;
    const int laneL = lane & 15;
    const int hi    = lane >> 4;

    const __bf16* arow = lhat + (long long)(m0 + laneL) * NN + hi * 16;
    const __bf16* br0  = xT + (long long)(n0 +  0 + laneL) * NN + hi * 16;
    const __bf16* br1  = xT + (long long)(n0 + 16 + laneL) * NN + hi * 16;
    const __bf16* br2  = xT + (long long)(n0 + 32 + laneL) * NN + hi * 16;
    const __bf16* br3  = xT + (long long)(n0 + 48 + laneL) * NN + hi * 16;

    v8f acc[4] = {};
#pragma unroll 2
    for (int k0 = 0; k0 < NN; k0 += 32) {
        v16bf af = cat16(*(const v8bf*)(arow + k0), *(const v8bf*)(arow + k0 + 8));
        v16bf b0 = cat16(*(const v8bf*)(br0 + k0),  *(const v8bf*)(br0 + k0 + 8));
        acc[0] = __builtin_amdgcn_wmma_f32_16x16x32_bf16(false, af, false, b0, (short)0, acc[0], false, false);
        v16bf b1 = cat16(*(const v8bf*)(br1 + k0),  *(const v8bf*)(br1 + k0 + 8));
        acc[1] = __builtin_amdgcn_wmma_f32_16x16x32_bf16(false, af, false, b1, (short)0, acc[1], false, false);
        v16bf b2 = cat16(*(const v8bf*)(br2 + k0),  *(const v8bf*)(br2 + k0 + 8));
        acc[2] = __builtin_amdgcn_wmma_f32_16x16x32_bf16(false, af, false, b2, (short)0, acc[2], false, false);
        v16bf b3 = cat16(*(const v8bf*)(br3 + k0),  *(const v8bf*)(br3 + k0 + 8));
        acc[3] = __builtin_amdgcn_wmma_f32_16x16x32_bf16(false, af, false, b3, (short)0, acc[3], false, false);
    }

#pragma unroll
    for (int t = 0; t < 4; ++t) {
        int n = n0 + t * 16 + laneL;
#pragma unroll
        for (int r = 0; r < 8; ++r) {
            int m = m0 + r + hi * 8;
            long long idx = (long long)m * FF + n;
            float v = x[idx] - acc[t][r];
            yhat[idx] = v;
            yhb[idx]  = (__bf16)v;
        }
    }
}

// ---------------------------------------------------------------------------
// Kernel 4: sn[n] = ||y_hat[n]||^2
// ---------------------------------------------------------------------------
__global__ __launch_bounds__(128) void sn_kernel(const float* __restrict__ yhat,
                                                 float* __restrict__ sn) {
    __shared__ float red[128];
    int row = blockIdx.x, t = threadIdx.x;
    const float* r = yhat + (long long)row * FF;
    float s = 0.0f;
    for (int f = t; f < FF; f += 128) { float v = r[f]; s += v * v; }
    red[t] = s; __syncthreads();
    for (int off = 64; off > 0; off >>= 1) {
        if (t < off) red[t] += red[t + off];
        __syncthreads();
    }
    if (t == 0) sn[row] = red[0];
}

// ---------------------------------------------------------------------------
// Kernel 5: Gram = yhb @ yhbT via WMMA; fused safe-sqrt + per-class atomics.
// block = 8 waves as 2(I) x 4(J) of 16x16 tiles -> 32 x 64 per block.
// yhb (4 MB) is L2-resident, direct global fragment loads.
// ---------------------------------------------------------------------------
__global__ __launch_bounds__(256) void gram_hl2_kernel(const __bf16* __restrict__ yhb,
                                                       const float* __restrict__ sn,
                                                       const unsigned char* __restrict__ mask,
                                                       const int* __restrict__ y,
                                                       float* __restrict__ num) {
    __shared__ float cls[NCLS];
    int tid = threadIdx.x, wave = tid >> 5, lane = tid & 31;
    if (tid < NCLS) cls[tid] = 0.0f;
    __syncthreads();

    int i0 = blockIdx.x * 32 + (wave >> 2) * 16;
    int j0 = blockIdx.y * 64 + (wave & 3) * 16;
    int laneL = lane & 15, hi = lane >> 4;

    const __bf16* aRow = yhb + (long long)(i0 + laneL) * FF + hi * 16;
    const __bf16* bRow = yhb + (long long)(j0 + laneL) * FF + hi * 16;  // B col n == Y row n

    v8f acc = {};
#pragma unroll 4
    for (int k0 = 0; k0 < FF; k0 += 32) {
        v16bf af = cat16(*(const v8bf*)(aRow + k0), *(const v8bf*)(aRow + k0 + 8));
        v16bf bf = cat16(*(const v8bf*)(bRow + k0), *(const v8bf*)(bRow + k0 + 8));
        acc = __builtin_amdgcn_wmma_f32_16x16x32_bf16(
            false, af, false, bf, (short)0, acc, false, false);
    }

    int j = j0 + laneL;
    bool mj = mask[j] != 0;
    int  yj = y[j];
#pragma unroll
    for (int r = 0; r < 8; ++r) {
        int i = i0 + r + hi * 8;
        float sq = sn[i] + sn[j] - 2.0f * acc[r];
        float d  = sq > 0.0f ? sqrtf(sq) : 0.0f;
        if (mj && mask[i] && y[i] == yj) atomicAdd(&cls[yj], d);
    }
    __syncthreads();
    if (tid < NCLS) { float v = cls[tid]; if (v != 0.0f) atomicAdd(&num[tid], v); }
}

// ---------------------------------------------------------------------------
// Kernel 6: hl1 — per-group mean pairwise distance among 7 nodes (f32, exact).
// ---------------------------------------------------------------------------
__global__ __launch_bounds__(64) void hl1_kernel(const float* __restrict__ yhat,
                                                 const int* __restrict__ groups,
                                                 float* __restrict__ hl1acc) {
    __shared__ float g[NCLS * FF];
    __shared__ float red[64];
    int grp = blockIdx.x, t = threadIdx.x;
    for (int i = 0; i < NCLS; ++i) {
        int node = groups[grp * NCLS + i];
        const float* r = yhat + (long long)node * FF;
        for (int f = t; f < FF; f += 64) g[i * FF + f] = r[f];
    }
    __syncthreads();
    float total = 0.0f;
    for (int i = 0; i < NCLS; ++i) {
        for (int j = i + 1; j < NCLS; ++j) {
            float s = 0.0f;
            for (int f = t; f < FF; f += 64) {
                float d = g[i * FF + f] - g[j * FF + f];
                s += d * d;
            }
            red[t] = s; __syncthreads();
            for (int off = 32; off > 0; off >>= 1) {
                if (t < off) red[t] += red[t + off];
                __syncthreads();
            }
            if (t == 0) { float sq = red[0]; total += (sq > 0.0f) ? sqrtf(sq) : 0.0f; }
            __syncthreads();
        }
    }
    if (t == 0) atomicAdd(hl1acc, 2.0f * total / 49.0f);  // both orders, /S^2
}

// ---------------------------------------------------------------------------
// Kernel 7: finalize scalar loss.
// ---------------------------------------------------------------------------
__global__ void finalize_kernel(const float* __restrict__ scal, float* __restrict__ out) {
    if (threadIdx.x == 0 && blockIdx.x == 0) {
        float hl2 = 0.0f;
        for (int c = 0; c < NCLS; ++c) {
            float cnt = scal[9 + c];
            if (cnt > 0.0f) hl2 += scal[2 + c] / fmaxf(cnt * cnt, 1.0f);
        }
        const float beta = (float)NG / (float)NCLS;
        out[0] = scal[0] + hl2 - scal[1] / beta;   // hl1 = -sum(means)
    }
}

// ---------------------------------------------------------------------------
extern "C" void kernel_launch(void* const* d_in, const int* in_sizes, int n_in,
                              void* d_out, int out_size, void* d_ws, size_t ws_size,
                              hipStream_t stream) {
    const float*         D      = (const float*)d_in[0];
    const float*         x      = (const float*)d_in[1];
    const float*         C      = (const float*)d_in[2];
    const unsigned char* mask   = (const unsigned char*)d_in[3];
    const int*           y      = (const int*)d_in[4];
    const int*           groups = (const int*)d_in[5];

    char* ws = (char*)d_ws;
    float*  scal = (float*)ws;                           // 64 floats
    float*  cn   = (float*)(ws + 256);                   // 11 floats
    float*  sn   = (float*)(ws + 1024);                  // 16 KB
    float*  yhat = (float*)(ws + (1ull << 20));          // 8 MB   @ 1 MB
    __bf16* yhb  = (__bf16*)(ws + (10ull << 20));        // 4 MB   @ 10 MB
    __bf16* xT   = (__bf16*)(ws + (15ull << 20));        // 4 MB   @ 15 MB (transposed x)
    __bf16* lhat = (__bf16*)(ws + (20ull << 20));        // 33.5 MB @ 20 MB

    prep_kernel<<<1, 256, 0, stream>>>(C, mask, y, scal, cn);

    lhat_kernel<<<(NN * (long long)NN) / (4 * 256), 256, 0, stream>>>(D, cn, lhat);

    xpose_kernel<<<dim3(NN / 32, FF / 32), 256, 0, stream>>>(x, xT);

    yhat_gemm_kernel<<<dim3(NN / 128, FF / 64), 256, 0, stream>>>(lhat, xT, x, yhat, yhb);

    sn_kernel<<<NN, 128, 0, stream>>>(yhat, sn);

    gram_hl2_kernel<<<dim3(NN / 32, NN / 64), 256, 0, stream>>>(yhb, sn, mask, y, scal + 2);

    hl1_kernel<<<NG, 64, 0, stream>>>(yhat, groups, scal + 1);

    finalize_kernel<<<1, 1, 0, stream>>>(scal, (float*)d_out);
}
